// Attention_17016660427353
// MI455X (gfx1250) — matrix operand order repaired and checked by reading, not compiled, not run on hardware
//
#include <hip/hip_runtime.h>
#include <math.h>

typedef __attribute__((ext_vector_type(16))) __bf16 v16bf;
typedef __attribute__((ext_vector_type(8)))  __bf16 v8bf;
typedef __attribute__((ext_vector_type(8)))  float  v8f;

constexpr int Bb = 4, Ss = 1024, Dd = 4096, Hh = 32, HDd = 128;

// ---------------------------------------------------------------------------
// GEMM: Y[m, n] = sum_k X[m, k] * W[n, k]   (X: M x 4096, W: 4096 x 4096)
// Double-buffered LDS ping-pong + global_prefetch three tiles ahead.
// OUT_MODE 0: f32 row-major output (M x 4096)           -> final projection
// OUT_MODE 1: bf16 (B,H,S,HD) output with fused RoPE    -> Q, K
// OUT_MODE 2: bf16 (B,H,S,HD) output, no RoPE           -> V
// ---------------------------------------------------------------------------
template <bool A_BF16, int OUT_MODE>
__global__ __launch_bounds__(256) void gemm_xwT(
    const void* __restrict__ Xv, const float* __restrict__ W,
    void* __restrict__ Outv,
    const float* __restrict__ fcos, const float* __restrict__ fsin)
{
  __shared__ alignas(32) __bf16 As[2][128][32];   // [buf][m][k]
  __shared__ alignas(32) __bf16 Bs[2][128][32];   // [buf][n][k]

  const int tid  = threadIdx.x;
  const int wave = tid >> 5;
  const int lane = tid & 31;
  const int hfl  = lane >> 4;      // which 16-lane half
  const int lr   = lane & 15;
  const int mtile = blockIdx.y * 128;
  const int ntile = blockIdx.x * 128;

  // register staging for the next tile
  float4 ra[4];
  v8bf   ra16[2];
  float4 rb[4];

  auto fetchA = [&](int kk) {
    if constexpr (A_BF16) {
      const __bf16* X = (const __bf16*)Xv;
#pragma unroll
      for (int r = 0; r < 2; ++r) {
        int idx = tid + 256 * r, row = idx >> 2, seg = idx & 3;
        ra16[r] = *(const v8bf*)(X + (size_t)(mtile + row) * Dd + kk + seg * 8);
      }
    } else {
      const float* X = (const float*)Xv;
#pragma unroll
      for (int r = 0; r < 4; ++r) {
        int idx = tid + 256 * r, row = idx >> 3, seg = idx & 7;
        ra[r] = *(const float4*)(X + (size_t)(mtile + row) * Dd + kk + seg * 4);
      }
    }
  };
  auto fetchB = [&](int kk) {
#pragma unroll
    for (int r = 0; r < 4; ++r) {
      int idx = tid + 256 * r, row = idx >> 3, seg = idx & 7;
      rb[r] = *(const float4*)(W + (size_t)(ntile + row) * Dd + kk + seg * 4);
    }
  };
  auto storeA = [&](int pb) {
    if constexpr (A_BF16) {
#pragma unroll
      for (int r = 0; r < 2; ++r) {
        int idx = tid + 256 * r, row = idx >> 2, seg = idx & 3;
        *(v8bf*)(&As[pb][row][seg * 8]) = ra16[r];
      }
    } else {
#pragma unroll
      for (int r = 0; r < 4; ++r) {
        int idx = tid + 256 * r, row = idx >> 3, seg = idx & 7;
        As[pb][row][seg * 4 + 0] = (__bf16)ra[r].x;
        As[pb][row][seg * 4 + 1] = (__bf16)ra[r].y;
        As[pb][row][seg * 4 + 2] = (__bf16)ra[r].z;
        As[pb][row][seg * 4 + 3] = (__bf16)ra[r].w;
      }
    }
  };
  auto storeB = [&](int pb) {
#pragma unroll
    for (int r = 0; r < 4; ++r) {
      int idx = tid + 256 * r, row = idx >> 3, seg = idx & 7;
      Bs[pb][row][seg * 4 + 0] = (__bf16)rb[r].x;
      Bs[pb][row][seg * 4 + 1] = (__bf16)rb[r].y;
      Bs[pb][row][seg * 4 + 2] = (__bf16)rb[r].z;
      Bs[pb][row][seg * 4 + 3] = (__bf16)rb[r].w;
    }
  };

  v8f c[8];
#pragma unroll
  for (int j = 0; j < 8; ++j) c[j] = v8f{};

  // prologue: tile 0 into buffer 0
  fetchA(0); fetchB(0);
  storeA(0); storeB(0);
  int p = 0;

  for (int k0 = 0; k0 < Dd; k0 += 32) {
    __syncthreads();
    const bool has_next = (k0 + 32) < Dd;
    if (has_next) {
      fetchA(k0 + 32);
      fetchB(k0 + 32);
      // prefetch three tiles ahead: one 128B row-chunk per thread covers the tile
      if (k0 + 96 < Dd) {
        if (tid < 128) {
          if constexpr (A_BF16)
            __builtin_prefetch((const __bf16*)Xv + (size_t)(mtile + tid) * Dd + k0 + 96, 0, 0);
          else
            __builtin_prefetch((const float*)Xv + (size_t)(mtile + tid) * Dd + k0 + 96, 0, 0);
        } else {
          __builtin_prefetch(W + (size_t)(ntile + tid - 128) * Dd + k0 + 96, 0, 0);
        }
      }
    }

    // ---- A fragment: lane = M row, elems = K hfl*16+0..15
    const int am = wave * 16 + lr;
    v16bf a;
    {
      v8bf a0 = *(const v8bf*)(&As[p][am][hfl * 16]);
      v8bf a1 = *(const v8bf*)(&As[p][am][hfl * 16 + 8]);
#pragma unroll
      for (int i = 0; i < 8; ++i) { a[i] = a0[i]; a[8 + i] = a1[i]; }
    }
    // ---- 8 WMMA tiles along N; B fragment: lane = N, elems = K hfl*16+0..15
#pragma unroll
    for (int j = 0; j < 8; ++j) {
      v16bf bf = *(const v16bf*)(&Bs[p][j * 16 + lr][hfl * 16]);
      c[j] = __builtin_amdgcn_wmma_f32_16x16x32_bf16(false, a, false, bf,
                                                     (short)0, c[j], false, false);
    }

    if (has_next) { storeA(p ^ 1); storeB(p ^ 1); }
    p ^= 1;
  }

  // -------- epilogue --------
  if (OUT_MODE == 0) {
    float* O = (float*)Outv;
#pragma unroll
    for (int j = 0; j < 8; ++j)
#pragma unroll
      for (int vv = 0; vv < 8; ++vv) {
        int row = mtile + wave * 16 + vv + 8 * hfl;
        int col = ntile + j * 16 + lr;
        O[(size_t)row * Dd + col] = c[j][vv];
      }
  } else {
    __bf16* O = (__bf16*)Outv;
#pragma unroll
    for (int j = 0; j < 8; ++j)
#pragma unroll
      for (int vv = 0; vv < 8; ++vv) {
        int m = mtile + wave * 16 + vv + 8 * hfl;
        int b = m >> 10, s = m & (Ss - 1);
        int e = ntile + j * 16 + lr;
        int h = e >> 7, dd = e & (HDd - 1);
        float val = c[j][vv];
        float res;
        if (OUT_MODE == 1) {
          // RoPE: pair elements are adjacent lanes (even = x0, odd = x1)
          float prt = __shfl_xor(val, 1, 32);
          float cs = fcos[s * (HDd / 2) + (dd >> 1)];
          float sn = fsin[s * (HDd / 2) + (dd >> 1)];
          res = ((lr & 1) == 0) ? (val * cs - prt * sn)
                                : (prt * sn + val * cs);
        } else {
          res = val;
        }
        O[(((size_t)(b * Hh + h)) * Ss + s) * HDd + dd] = (__bf16)res;
      }
  }
}

// ---------------------------------------------------------------------------
// Flash attention: per (b,h), q-tile of 128 rows, key tiles of 64, causal.
// K tile staged via GLOBAL_LOAD_ASYNC_TO_LDS_B128 (ASYNCcnt); V transposed
// through registers. q/k/v bf16 in (B,H,S,HD); output bf16 in (B,S,D).
// ---------------------------------------------------------------------------
__global__ __launch_bounds__(256) void attn_kernel(
    const __bf16* __restrict__ q, const __bf16* __restrict__ k,
    const __bf16* __restrict__ v, __bf16* __restrict__ out)
{
  __shared__ alignas(32) __bf16 Ks[64][128];      // [key][d]
  __shared__ alignas(32) __bf16 Vt[128][64];      // [d][key]
  __shared__ alignas(32) __bf16 Ps[8][16][64];    // per-wave probs [m][key]

  const int tid  = threadIdx.x;
  const int wave = tid >> 5;
  const int lane = tid & 31;
  const int hfl  = lane >> 4;
  const int lr   = lane & 15;
  const int qt   = blockIdx.x;        // 0..7
  const int bh   = blockIdx.y;        // 0..127
  const size_t base = (size_t)bh * Ss * HDd;
  const __bf16* qp = q + base;
  const __bf16* kp = k + base;
  const __bf16* vp = v + base;

  // Q fragments for this wave's 16 rows, HD split into 4 K-chunks of 32
  v16bf qa[4];
  {
    const int qrow = qt * 128 + wave * 16 + lr;
#pragma unroll
    for (int kc = 0; kc < 4; ++kc) {
      v8bf a0 = *(const v8bf*)(qp + (size_t)qrow * HDd + kc * 32 + hfl * 16);
      v8bf a1 = *(const v8bf*)(qp + (size_t)qrow * HDd + kc * 32 + hfl * 16 + 8);
#pragma unroll
      for (int i = 0; i < 8; ++i) { qa[kc][i] = a0[i]; qa[kc][8 + i] = a1[i]; }
    }
  }

  float mrow[8], lrow[8];
  v8f oacc[8];
#pragma unroll
  for (int vv = 0; vv < 8; ++vv) { mrow[vv] = -__builtin_inff(); lrow[vv] = 0.f; }
#pragma unroll
  for (int j = 0; j < 8; ++j) oacc[j] = v8f{};

  const float scale = 0.08838834764831845f;   // 1/sqrt(128)
  const int nkt = 2 * qt + 2;                 // causal: key tiles of 64

  for (int kt = 0; kt < nkt; ++kt) {
    __syncthreads();
    // ---- stage K tile (64 x 128, raw copy) via async load-to-LDS ----
#pragma unroll
    for (int r = 0; r < 4; ++r) {
      int idx = tid + 256 * r;                // 0..1023, 16B granules
      int row = idx >> 4, seg = idx & 15;
      unsigned int lds = (unsigned int)(uintptr_t)(&Ks[row][seg * 8]);
      unsigned long long ga =
          (unsigned long long)(uintptr_t)(kp + (size_t)(kt * 64 + row) * HDd + seg * 8);
      asm volatile("global_load_async_to_lds_b128 %0, %1, off"
                   :: "v"(lds), "v"(ga) : "memory");
    }
    // ---- V tile transposed through registers (overlaps with async K) ----
#pragma unroll
    for (int r = 0; r < 4; ++r) {
      int idx = tid + 256 * r;
      int row = idx >> 4, seg = idx & 15;
      v8bf vv8 = *(const v8bf*)(vp + (size_t)(kt * 64 + row) * HDd + seg * 8);
#pragma unroll
      for (int i = 0; i < 8; ++i) Vt[seg * 8 + i][row] = vv8[i];
    }
    asm volatile("s_wait_asynccnt 0" ::: "memory");
    __syncthreads();

    // ---- scores: S = Q . K^T  (4 key blocks of 16) ----
    v8f sc[4];
#pragma unroll
    for (int kb = 0; kb < 4; ++kb) {
      v8f s = v8f{};
      const int kl = kb * 16 + lr;
#pragma unroll
      for (int kc = 0; kc < 4; ++kc) {
        v16bf bf = *(const v16bf*)(&Ks[kl][kc * 32 + hfl * 16]);
        s = __builtin_amdgcn_wmma_f32_16x16x32_bf16(false, qa[kc], false, bf,
                                                    (short)0, s, false, false);
      }
      sc[kb] = s;
    }

    // ---- causal mask + online softmax, write P to LDS ----
#pragma unroll
    for (int vv = 0; vv < 8; ++vv) {
      const int qg = qt * 128 + wave * 16 + vv + 8 * hfl;
      float sv[4];
      float rmax = -__builtin_inff();
#pragma unroll
      for (int kb = 0; kb < 4; ++kb) {
        int kg = kt * 64 + kb * 16 + lr;
        float x = sc[kb][vv] * scale;
        if (kg > qg) x -= 1e9f;               // == mask add of -1e9
        sv[kb] = x;
        rmax = fmaxf(rmax, x);
      }
#pragma unroll
      for (int msk = 1; msk < 16; msk <<= 1)
        rmax = fmaxf(rmax, __shfl_xor(rmax, msk, 32));
      float mnew  = fmaxf(mrow[vv], rmax);
      float alpha = __expf(mrow[vv] - mnew);
      float rsum = 0.f;
#pragma unroll
      for (int kb = 0; kb < 4; ++kb) {
        float pr = __expf(sv[kb] - mnew);
        rsum += pr;
        Ps[wave][vv + 8 * hfl][kb * 16 + lr] = (__bf16)pr;
      }
#pragma unroll
      for (int msk = 1; msk < 16; msk <<= 1)
        rsum += __shfl_xor(rsum, msk, 32);
      lrow[vv] = lrow[vv] * alpha + rsum;
      mrow[vv] = mnew;
#pragma unroll
      for (int j = 0; j < 8; ++j) oacc[j][vv] *= alpha;
    }
    __syncthreads();

    // ---- O += P . V  (keys split into 2 chunks of 32) ----
#pragma unroll
    for (int kc2 = 0; kc2 < 2; ++kc2) {
      v16bf pa;
      {
        v8bf p0 = *(const v8bf*)(&Ps[wave][lr][kc2 * 32 + hfl * 16]);
        v8bf p1 = *(const v8bf*)(&Ps[wave][lr][kc2 * 32 + hfl * 16 + 8]);
#pragma unroll
        for (int i = 0; i < 8; ++i) { pa[i] = p0[i]; pa[8 + i] = p1[i]; }
      }
#pragma unroll
      for (int j = 0; j < 8; ++j) {
        v16bf bf = *(const v16bf*)(&Vt[j * 16 + lr][kc2 * 32 + hfl * 16]);
        oacc[j] = __builtin_amdgcn_wmma_f32_16x16x32_bf16(false, pa, false, bf,
                                                          (short)0, oacc[j],
                                                          false, false);
      }
    }
  }

  // ---- normalize and store to (B,S,D) bf16 ----
  const int b = bh >> 5, h = bh & 31;
#pragma unroll
  for (int j = 0; j < 8; ++j)
#pragma unroll
    for (int vv = 0; vv < 8; ++vv) {
      int tq = qt * 128 + wave * 16 + vv + 8 * hfl;
      int e  = h * HDd + j * 16 + lr;
      float o = oacc[j][vv] / lrow[vv];
      out[((size_t)(b * Ss) + tq) * Dd + e] = (__bf16)o;
    }
}

// ---------------------------------------------------------------------------
extern "C" void kernel_launch(void* const* d_in, const int* in_sizes, int n_in,
                              void* d_out, int out_size, void* d_ws, size_t ws_size,
                              hipStream_t stream) {
  const float* x    = (const float*)d_in[0];
  const float* wq   = (const float*)d_in[1];
  const float* wk   = (const float*)d_in[2];
  const float* wv   = (const float*)d_in[3];
  const float* wo   = (const float*)d_in[4];
  const float* fcos = (const float*)d_in[7];
  const float* fsin = (const float*)d_in[8];

  // workspace: q | k | v | attn_out, all bf16 (32 MB each, 128 MB total)
  const size_t QKV = (size_t)Bb * Hh * Ss * HDd;
  __bf16* qws = (__bf16*)d_ws;
  __bf16* kws = qws + QKV;
  __bf16* vws = kws + QKV;
  __bf16* aws = vws + QKV;

  dim3 blk(256);
  dim3 gproj(Dd / 128, (Bb * Ss) / 128);        // 32 x 32
  gemm_xwT<false, 1><<<gproj, blk, 0, stream>>>(x, wq, qws, fcos, fsin);
  gemm_xwT<false, 1><<<gproj, blk, 0, stream>>>(x, wk, kws, fcos, fsin);
  gemm_xwT<false, 2><<<gproj, blk, 0, stream>>>(x, wv, vws, fcos, fsin);

  dim3 gattn(Ss / 128, Bb * Hh);                // 8 x 128
  attn_kernel<<<gattn, blk, 0, stream>>>(qws, kws, vws, aws);

  gemm_xwT<true, 0><<<gproj, blk, 0, stream>>>(aws, wo, d_out, nullptr, nullptr);
}
